// MinConv2dGRU_20916490731576
// MI455X (gfx1250) — matrix operand order repaired and checked by reading, not compiled, not run on hardware
//
#include <hip/hip_runtime.h>
#include <hip/hip_bf16.h>

// MinConv2dGRU for MI455X (gfx1250):
//   - f16 WMMA implicit-GEMM 3x3 convs (v_wmma_f32_16x16x32_f16, f32 accum)
//   - fragment-order weight packing (1 contiguous 32B load per A fragment)
//   - NHWC-f16 activations so B fragments are contiguous 32B LDS loads
//   - async global->LDS staging (global_load_async_to_lds_b128 / s_wait_asynccnt)
//   - fp32 minGRU scan (bandwidth-bound elementwise, 16 steps)
// B=4, S=16, CIN=32, H=W=64, HID=64.

typedef __attribute__((ext_vector_type(16))) _Float16 v16h;
typedef __attribute__((ext_vector_type(8)))  float    v8f;

#define NIMG 64          // B*S
#define PIX  4096        // H*W
#define HH   64
#define WW   64

// ---------------------------------------------------------------------------
// NCHW fp32 -> NHWC f16  (for conv input staging); CIN compile-time.
// ---------------------------------------------------------------------------
template <int CIN, int LGC>
__global__ __launch_bounds__(256)
void nchw_to_nhwc_f16(const float* __restrict__ in, _Float16* __restrict__ outp)
{
    int gidx = blockIdx.x * 256 + threadIdx.x;       // NIMG*CIN*PIX exact
    int p   = gidx & 4095;
    int c   = (gidx >> 12) & (CIN - 1);
    int img = gidx >> (12 + LGC);
    outp[((size_t)img * PIX + p) * CIN + c] = (_Float16)in[gidx];
}

// ---------------------------------------------------------------------------
// Pack fp32 weights into f16 WMMA-A *fragment order*:
//   wcp[mt][kc][lane][slot]  (slot 0..15, 32 bytes per lane per k-chunk)
//   lane = lo + 16*hi ; slot s -> kkA = hi*16 + s
//   k = kc*32 + kkA ;  k = (kh*3+kw)*CIN + ic
// Grid: (NKC, NMT), block 512 (= 32 lanes x 16 slots, tid = lane*16+slot).
// Layer0 also appends R0 rows (oc 128..191) after W0 (oc 0..127).
// ---------------------------------------------------------------------------
template <int CIN, int LGC, int NKC, int SPLIT>   // SPLIT: oc >= SPLIT -> Wb
__global__ __launch_bounds__(512)
void pack_weights(const float* __restrict__ Wa, const float* __restrict__ Wb,
                  _Float16* __restrict__ wcp)
{
    const int kc  = blockIdx.x;
    const int mt  = blockIdx.y;
    const int tid = threadIdx.x;
    const int slot = tid & 15;
    const int lane = tid >> 4;
    const int lo = lane & 15, hi = lane >> 4;
    const int m  = mt * 16 + lo;
    const int kk = hi * 16 + slot;
    const int k  = kc * 32 + kk;
    const int tap = k >> LGC;                 // kh*3+kw
    const int ic  = k & (CIN - 1);
    const int KTOT = 9 * CIN;
    float v = (m < SPLIT) ? Wa[(size_t)m * KTOT + ic * 9 + tap]
                          : Wb[(size_t)(m - SPLIT) * KTOT + ic * 9 + tap];
    wcp[(size_t)(mt * NKC + kc) * 512 + tid] = (_Float16)v;
}

// ---------------------------------------------------------------------------
// Implicit-GEMM 3x3 conv (pad=1) via v_wmma_f32_16x16x32_f16.
// Grid: (H rows, NIMG images).  Block: 256 threads = 8 waves.
// LDS: s_in[kh 0..2][col 0..65][ic]  f16, halo columns zeroed.
// Per k-chunk: A = one 32B contiguous global load (fragment-packed weights),
//              B = one 32B contiguous LDS load (channels-innermost layout).
// yl[img][oc][pix] fp32, bias added for oc < 128.
// ---------------------------------------------------------------------------
template <int CIN, int LGC, int NOC, int K>
__global__ __launch_bounds__(256)
void conv_wmma(const _Float16* __restrict__ xh,
               const _Float16* __restrict__ wcp,
               const float* __restrict__ bias, float* __restrict__ yl)
{
    const int h   = blockIdx.x;
    const int img = blockIdx.y;
    const int tid = threadIdx.x;

    __shared__ __align__(32) _Float16 s_in[3 * 66 * CIN];

    // Zero halo columns (col 0 and col 65) for all 3 kh rows.
    #pragma unroll
    for (int base = 0; base < 3 * 2 * CIN; base += 256) {
        int idx = base + tid;
        if (idx < 3 * 2 * CIN) {
            int kh   = idx >> (LGC + 1);
            int r    = idx & (2 * CIN - 1);
            int side = (r >= CIN) ? 65 : 0;
            int ic   = r & (CIN - 1);
            s_in[(kh * 66 + side) * CIN + ic] = (_Float16)0.f;
        }
    }

    // Stage 3 input rows: contiguous NHWC rows -> LDS via async copy.
    constexpr int rowHalves = WW * CIN;                 // 2048 or 4096
    constexpr int nchunk    = rowHalves >> 3;           // 16B chunks: 256 or 512
    #pragma unroll
    for (int kh = 0; kh < 3; ++kh) {
        int gh = h + kh - 1;
        _Float16* dst = &s_in[(kh * 66 + 1) * CIN];
        if (gh >= 0 && gh < HH) {
            const _Float16* src = xh + ((size_t)img * HH + gh) * rowHalves;
            #pragma unroll
            for (int cb = 0; cb < nchunk; cb += 256) {
                unsigned lds_off = (unsigned)(uintptr_t)(dst + (cb + tid) * 8);
                const _Float16* g = src + (cb + tid) * 8;
                asm volatile("global_load_async_to_lds_b128 %0, %1, off"
                             :: "v"(lds_off), "v"(g) : "memory");
            }
        } else {
            #pragma unroll
            for (int i2 = 0; i2 < rowHalves; i2 += 256)
                dst[i2 + tid] = (_Float16)0.f;
        }
    }
    asm volatile("s_wait_asynccnt 0x0" ::: "memory");
    __syncthreads();

    const int wave = tid >> 5;
    const int lane = tid & 31;
    const int lo   = lane & 15;
    const int hi   = lane >> 4;
    constexpr int NMT    = NOC >> 4;
    constexpr int NTILES = NMT * 4;
    constexpr int NKC    = K >> 5;

    #pragma unroll 1
    for (int tile = wave; tile < NTILES; tile += 8) {
        const int mt  = tile >> 2;
        const int pt  = tile & 3;
        const int px0 = pt << 4;
        const _Float16* wbase = wcp + ((size_t)mt * NKC * 32 + lane) * 16;

        v8f acc = {};
        #pragma unroll
        for (int kc = 0; kc < NKC; ++kc) {
            v16h amat = *(const v16h*)(wbase + (size_t)kc * 512);

            // B fragment: one contiguous 16-half run, k = kc*32 + hi*16 + [0..15]
            const int kb  = kc * 32 + hi * 16;
            const int tap = kb >> LGC;           // run never crosses a tap
            const int icb = kb & (CIN - 1);
            const int kh  = (tap >= 6) ? 2 : ((tap >= 3) ? 1 : 0);
            const int kw  = tap - 3 * kh;
            const int col = px0 + lo + kw;       // padded col (halo at 0/65)
            v16h bmat = *(const v16h*)&s_in[(kh * 66 + col) * CIN + icb];

            acc = __builtin_amdgcn_wmma_f32_16x16x32_f16(
                      false, amat, false, bmat, (short)0, acc, false, false);
        }

        // D layout: VGPR r -> (M = mt*16 + r + 8*hi, N = px0 + lo).
        #pragma unroll
        for (int r = 0; r < 8; ++r) {
            int oc = (mt << 4) + r + 8 * hi;
            int p  = (h << 6) + px0 + lo;
            float v = acc[r];
            if (oc < 128) v += bias[oc];
            yl[((size_t)img * NOC + oc) * PIX + p] = v;
        }
    }
}

// ---------------------------------------------------------------------------
// minGRU scan over S=16 + residual add.
//   h_t = sigmoid(g)*g~(hd) + (1-sigmoid(g))*h_{t-1},  h_0 = 0.5
//   g~(x) = x+0.5 (x>=0) else sigmoid(x)
// One thread per (b, c, h, w).  Writes scan+res to outm (fp32), optional
// f16 NHWC copy to outh (feeds next conv), final pre-residual h to hout.
// ---------------------------------------------------------------------------
__global__ __launch_bounds__(256)
void mingru_scan(const float* __restrict__ yl, int noc,
                 const float* __restrict__ res, int res_ocs, int res_cb,
                 float* __restrict__ outm, _Float16* __restrict__ outh,
                 float* __restrict__ hout)
{
    int gidx = blockIdx.x * 256 + threadIdx.x;     // 4 * 64 * 4096 threads
    int p  = gidx & 4095;
    int c  = (gidx >> 12) & 63;
    int bb = gidx >> 18;

    float hstate = 0.5f;
    #pragma unroll 4
    for (int s = 0; s < 16; ++s) {
        int img = bb * 16 + s;
        size_t base = (size_t)img * noc * PIX;
        float g  = yl[base + ((size_t)c << 12) + p];
        float hd = yl[base + ((size_t)(64 + c) << 12) + p];
        float z  = 1.f / (1.f + __expf(-g));
        float gg = (hd >= 0.f) ? (hd + 0.5f) : 1.f / (1.f + __expf(-hd));
        hstate = z * gg + (1.f - z) * hstate;
        float rv = res[((size_t)img * res_ocs + res_cb + c) * PIX + p];
        float ov = hstate + rv;
        outm[((size_t)img << 18) + ((size_t)c << 12) + p] = ov;
        if (outh)
            outh[((size_t)img * PIX + p) * 64 + c] = (_Float16)ov;
    }
    hout[((size_t)(bb * 64 + c) << 12) + p] = hstate;   // pre-residual last step
}

// ---------------------------------------------------------------------------
extern "C" void kernel_launch(void* const* d_in, const int* in_sizes, int n_in,
                              void* d_out, int out_size, void* d_ws, size_t ws_size,
                              hipStream_t stream)
{
    const float* x  = (const float*)d_in[0];
    const float* W0 = (const float*)d_in[1];
    const float* b0 = (const float*)d_in[2];
    const float* R0 = (const float*)d_in[3];
    const float* W1 = (const float*)d_in[4];
    const float* b1 = (const float*)d_in[5];
    float* out = (float*)d_out;

    char* ws = (char*)d_ws;
    _Float16* wcp0 = (_Float16*)ws;                          // 110592 B
    _Float16* wcp1 = (_Float16*)(ws + 110592);               // 147456 B
    _Float16* xh   = (_Float16*)(ws + 262144);               // 16 MB (NHWC f16 x)
    _Float16* o0h  = (_Float16*)(ws + 262144 + 16777216);    // 32 MB (NHWC f16 out0)
    float*    out0 = (float*)(ws + 262144 + 16777216 + 33554432);            // 64 MB
    float*    ylin = (float*)(ws + 262144 + 16777216 + 33554432 + 67108864); // 192 MB

    // outputs: out [16777216] | h0 [1048576] | h1 [1048576]
    float* h0_out = out + 16777216;
    float* h1_out = h0_out + 1048576;

    // Weight packing: grid (kc, mt), block 512 (lane*16+slot == tid).
    pack_weights<32, 5, 9, 128><<<dim3(9, 12), 512, 0, stream>>>(W0, R0, wcp0);
    pack_weights<64, 6, 18, 128><<<dim3(18, 8), 512, 0, stream>>>(W1, W1, wcp1);

    nchw_to_nhwc_f16<32, 5><<<(NIMG * 32 * PIX) / 256, 256, 0, stream>>>(x, xh);

    dim3 cgrid(HH, NIMG);
    // Layer 0: OC=192 (gate 0..63 | hidden 64..127 | residual 128..191), K=288.
    conv_wmma<32, 5, 192, 288><<<cgrid, 256, 0, stream>>>(xh, wcp0, b0, ylin);
    mingru_scan<<<(4 * 64 * PIX) / 256, 256, 0, stream>>>(
        ylin, 192, /*res=*/ylin, /*res_ocs=*/192, /*res_cb=*/128,
        out0, o0h, h0_out);

    // Layer 1: OC=128, K=576, residual = out0.
    conv_wmma<64, 6, 128, 576><<<cgrid, 256, 0, stream>>>(o0h, wcp1, b1, ylin);
    mingru_scan<<<(4 * 64 * PIX) / 256, 256, 0, stream>>>(
        ylin, 128, /*res=*/out0, /*res_ocs=*/64, /*res_cb=*/0,
        out, (_Float16*)nullptr, h1_out);
}
